// MiMoV2Attention_7172595384908
// MI455X (gfx1250) — matrix operand order repaired and checked by reading, not compiled, not run on hardware
//
#include <hip/hip_runtime.h>
#include <hip/hip_bf16.h>

// ---------------------------------------------------------------------------
// MiMoV2 attention for MI455X (gfx1250, wave32, WMMA).
// All GEMM-shaped work uses v_wmma_f32_16x16x32_bf16 (f32 accumulate).
// GEMMs: 128x128 block tile, 8 waves (2x4), 64x32 wave tile (4x2 frags),
// BK=32, LDS double-buffered, compile-time N/K for immediate-offset stores.
// ---------------------------------------------------------------------------

typedef __bf16 v16bf __attribute__((ext_vector_type(16)));
typedef float  v8f   __attribute__((ext_vector_type(8)));

union FragAB { v16bf v; unsigned u[8]; };
union FragC  { v8f   v; float    f[8]; };

static __device__ __forceinline__ __bf16 f2bf(float x) {
  unsigned u = __builtin_bit_cast(unsigned, x);
  unsigned short s = (unsigned short)((u + 0x7FFFu + ((u >> 16) & 1u)) >> 16);
  return __builtin_bit_cast(__bf16, s);
}

// ---------------------------------------------------------------------------
// 1) f32 -> bf16 convert (grid-stride)
// ---------------------------------------------------------------------------
__global__ void cvt_bf16_kernel(const float* __restrict__ src,
                                __bf16* __restrict__ dst, int n) {
  int i = blockIdx.x * blockDim.x + threadIdx.x;
  int stride = gridDim.x * blockDim.x;
  for (; i < n; i += stride) dst[i] = f2bf(src[i]);
}

// ---------------------------------------------------------------------------
// 2) bf16 GEMM: C[M,N](f32) = A[M,K](bf16,row) * B[K,N](bf16,row)
//    N, K compile-time; M covered by gridDim.y * 128.
// ---------------------------------------------------------------------------
template <int N, int K>
__global__ __launch_bounds__(256) void gemm_bf16_kernel(
    const __bf16* __restrict__ A, const __bf16* __restrict__ B,
    float* __restrict__ C) {
  constexpr int BM = 128, BN = 128, BK = 32;
  __shared__ __bf16 As[2][BM][BK];   // [m][k]
  __shared__ __bf16 Bs[2][BN][BK];   // transposed: [n][k]

  const int tid  = threadIdx.x;
  const int lane = tid & 31, wid = tid >> 5;
  const int wm = wid >> 2, wn = wid & 3;        // 2x4 wave grid
  const int bm = blockIdx.y * BM, bn = blockIdx.x * BN;
  const int l15 = lane & 15, hi = lane >> 4;

  const unsigned* A32 = (const unsigned*)A;
  const unsigned* B32 = (const unsigned*)B;

  // per-thread global load coordinates
  const int arow = tid >> 1, acd = (tid & 1) * 8;   // A: 8 dwords/thread
  const int bk   = (tid >> 4) * 2;                  // B: k-row pair
  const int bdw  = (tid & 15) * 4;                  // B: dword col base

  const unsigned* aptr  = A32 + (((size_t)(bm + arow) * K) >> 1) + acd;
  const unsigned* bptr0 = B32 + (((size_t)bk * N + bn) >> 1) + bdw;
  const unsigned* bptr1 = B32 + (((size_t)(bk + 1) * N + bn) >> 1) + bdw;

  unsigned areg[8], b0r[4], b1r[4];

  auto gload = [&](int k0) {
    const unsigned* ap = aptr + (k0 >> 1);
#pragma unroll
    for (int i = 0; i < 8; ++i) areg[i] = ap[i];
    const unsigned* bp0 = bptr0 + (size_t)k0 * (N / 2);
    const unsigned* bp1 = bptr1 + (size_t)k0 * (N / 2);
#pragma unroll
    for (int i = 0; i < 4; ++i) { b0r[i] = bp0[i]; b1r[i] = bp1[i]; }
  };
  auto lstore = [&](int bufv) {
    unsigned* ad = (unsigned*)&As[bufv][arow][0] + acd;
#pragma unroll
    for (int i = 0; i < 8; ++i) ad[i] = areg[i];
    unsigned* bd = (unsigned*)&Bs[bufv][0][0];
#pragma unroll
    for (int i = 0; i < 4; ++i) {
      int n = (bdw + i) * 2;
      unsigned t0 = (b0r[i] & 0xFFFFu) | (b1r[i] << 16);        // [n][k,k+1]
      unsigned t1 = (b0r[i] >> 16) | (b1r[i] & 0xFFFF0000u);    // [n+1][k,k+1]
      bd[n * (BK / 2) + (bk >> 1)]       = t0;
      bd[(n + 1) * (BK / 2) + (bk >> 1)] = t1;
    }
  };

  FragC acc[4][2];
#pragma unroll
  for (int i = 0; i < 4; ++i)
#pragma unroll
    for (int j = 0; j < 2; ++j)
#pragma unroll
      for (int e = 0; e < 8; ++e) acc[i][j].f[e] = 0.0f;

  gload(0);
  lstore(0);
  __syncthreads();

  int buf = 0;
  for (int k0 = 0; k0 < K; k0 += BK) {
    const bool has_next = (k0 + BK) < K;
    if (has_next) {
      gload(k0 + BK);
      if (k0 + 2 * BK < K) {   // gfx1250 global_prefetch_b8 for the k+2 tile
        __builtin_prefetch(aptr + ((k0 + 2 * BK) >> 1), 0, 1);
        __builtin_prefetch(bptr0 + (size_t)(k0 + 2 * BK) * (N / 2), 0, 1);
      }
    }

    FragAB af[4], bfr[2];
#pragma unroll
    for (int i = 0; i < 4; ++i) {
      const unsigned* ar = (const unsigned*)&As[buf][wm * 64 + i * 16 + l15][0];
#pragma unroll
      for (int r = 0; r < 8; ++r) {
        int kp = ((r & 4) ? 16 : 0) + (r & 3) * 2 + hi * 8;   // A frag K index
        af[i].u[r] = ar[kp >> 1];
      }
    }
#pragma unroll
    for (int j = 0; j < 2; ++j) {
      const unsigned* br = (const unsigned*)&Bs[buf][wn * 32 + j * 16 + l15][0];
#pragma unroll
      for (int r = 0; r < 8; ++r) bfr[j].u[r] = br[((r & 4) ? 8 : 0) + (r & 3) + hi * 4];  // K=(r&4?16:0)+hi*8+2(r&3)
    }
#pragma unroll
    for (int i = 0; i < 4; ++i)
#pragma unroll
      for (int j = 0; j < 2; ++j)
        acc[i][j].v = __builtin_amdgcn_wmma_f32_16x16x32_bf16(
            false, af[i].v, false, bfr[j].v, (short)0, acc[i][j].v, false, false);

    if (has_next) {
      lstore(buf ^ 1);
      __syncthreads();
      buf ^= 1;
    }
  }

  // epilogue: single base + compile-time immediate offsets
  float* cbase = C + (size_t)(bm + wm * 64 + hi * 8) * N + bn + wn * 32 + l15;
#pragma unroll
  for (int i = 0; i < 4; ++i)
#pragma unroll
    for (int j = 0; j < 2; ++j)
#pragma unroll
      for (int e = 0; e < 8; ++e)
        cbase[(i * 16 + e) * N + j * 16] = acc[i][j].f[e];
}

// ---------------------------------------------------------------------------
// 3a) RoPE + scale + pack: src f32 [S, nh*128] -> dst bf16 [nh][S][128]
// ---------------------------------------------------------------------------
__global__ void pack_rope_kernel(const float* __restrict__ src,
                                 const float* __restrict__ cosT,
                                 const float* __restrict__ sinT,
                                 __bf16* __restrict__ dst, int nh, float scale) {
  const int total = 2048 * nh * 128;
  int idx = blockIdx.x * blockDim.x + threadIdx.x;
  int stride = gridDim.x * blockDim.x;
  for (; idx < total; idx += stride) {
    int d  = idx & 127;
    int hs = idx >> 7;
    int s  = hs & 2047;
    int h  = hs >> 11;
    const float* row = src + (size_t)s * (nh * 128) + h * 128;
    float val;
    if (d < 64) {
      float x = row[d];
      float c = cosT[s * 64 + d], sn = sinT[s * 64 + d];
      float rot = (d < 32) ? -row[d + 32] : row[d - 32];
      val = x * c + rot * sn;
    } else {
      val = row[d];
    }
    dst[idx] = f2bf(val * scale);
  }
}

// 3b) V pack transposed: src f32 [S, 8*128] -> dst bf16 [8][128][S]
__global__ void pack_vT_kernel(const float* __restrict__ src,
                               __bf16* __restrict__ dst) {
  const int total = 8 * 128 * 2048;
  int idx = blockIdx.x * blockDim.x + threadIdx.x;
  int stride = gridDim.x * blockDim.x;
  for (; idx < total; idx += stride) {
    int s = idx & 2047;
    int t = idx >> 11;
    int d = t & 127;
    int h = t >> 7;
    dst[idx] = f2bf(src[(size_t)s * 1024 + h * 128 + d]);
  }
}

// ---------------------------------------------------------------------------
// 4) Flash attention with sink. One wave per (head, 16-query tile).
//    qb [32][2048][128], kb [8][2048][128], vt [8][128][2048], out [2048][4096]
// ---------------------------------------------------------------------------
__global__ __launch_bounds__(256) void flash_attn_kernel(
    const __bf16* __restrict__ qb, const __bf16* __restrict__ kb,
    const __bf16* __restrict__ vt, const float* __restrict__ sinks,
    __bf16* __restrict__ out) {
  __shared__ __bf16 pbuf[8][16][34];   // per-wave P tile (16x32, padded)

  const int lane = threadIdx.x & 31, wid = threadIdx.x >> 5;
  const int gid  = blockIdx.x * 8 + wid;     // 4096 waves total
  const int h    = gid >> 7;                 // 128 q-tiles per head
  const int qt   = gid & 127;
  const int kvh  = h >> 2;                   // GQA group of 4
  const int l15  = lane & 15, hi = lane >> 4;

  // Q fragments: 16x128 -> 4 A-frags of 16x32 (scaling already folded in)
  FragAB qf[4];
  {
    const unsigned* qrow =
        (const unsigned*)(qb + ((size_t)h * 2048 + qt * 16 + l15) * 128);
#pragma unroll
    for (int kc = 0; kc < 4; ++kc)
#pragma unroll
      for (int r = 0; r < 8; ++r) {
        int kp = kc * 32 + ((r & 4) ? 16 : 0) + (r & 3) * 2 + hi * 8;
        qf[kc].u[r] = qrow[kp >> 1];
      }
  }

  const float snk = sinks[h];
  float run_max[8], run_sum[8];
  FragC o[8];
#pragma unroll
  for (int e = 0; e < 8; ++e) { run_max[e] = snk; run_sum[e] = 1.0f; }
#pragma unroll
  for (int dt = 0; dt < 8; ++dt)
#pragma unroll
    for (int e = 0; e < 8; ++e) o[dt].f[e] = 0.0f;

  const int qrow_base = qt * 16;
  const int jmax = (qrow_base + 15) >> 5;

  for (int j = 0; j <= jmax; ++j) {
    // ---- scores: 16x32 tile via 8 WMMAs -------------------------------
    FragC sc[2];
#pragma unroll
    for (int nt = 0; nt < 2; ++nt) {
#pragma unroll
      for (int e = 0; e < 8; ++e) sc[nt].f[e] = 0.0f;
      int key = j * 32 + nt * 16 + l15;
      const unsigned* krow =
          (const unsigned*)(kb + ((size_t)kvh * 2048 + key) * 128);
#pragma unroll
      for (int kc = 0; kc < 4; ++kc) {
        FragAB bk;
#pragma unroll
        for (int r = 0; r < 8; ++r) bk.u[r] = krow[kc * 16 + ((r & 4) ? 8 : 0) + (r & 3) + hi * 4];
        sc[nt].v = __builtin_amdgcn_wmma_f32_16x16x32_bf16(
            false, qf[kc].v, false, bk.v, (short)0, sc[nt].v, false, false);
      }
    }
    // ---- online softmax (sink folded into init state) -----------------
#pragma unroll
    for (int e = 0; e < 8; ++e) {
      int row = qrow_base + e + hi * 8;
      int k0i = j * 32 + l15;
      float s0 = (k0i > row)      ? -1e30f : sc[0].f[e];
      float s1 = (k0i + 16 > row) ? -1e30f : sc[1].f[e];
      float tmax = fmaxf(s0, s1);
#pragma unroll
      for (int m = 1; m < 16; m <<= 1)
        tmax = fmaxf(tmax, __shfl_xor(tmax, m, 32));
      float nmax  = fmaxf(run_max[e], tmax);
      float alpha = __expf(run_max[e] - nmax);
      float p0 = __expf(s0 - nmax);
      float p1 = __expf(s1 - nmax);
      float ps = p0 + p1;
#pragma unroll
      for (int m = 1; m < 16; m <<= 1) ps += __shfl_xor(ps, m, 32);
      run_sum[e] = run_sum[e] * alpha + ps;
      run_max[e] = nmax;
#pragma unroll
      for (int dt = 0; dt < 8; ++dt) o[dt].f[e] *= alpha;
      int prow = e + hi * 8;
      pbuf[wid][prow][l15]      = f2bf(p0);
      pbuf[wid][prow][16 + l15] = f2bf(p1);
    }
    // ---- P: C-layout -> A-frag layout via per-wave LDS tile -----------
    FragAB pf;
    {
      const unsigned* prow = (const unsigned*)&pbuf[wid][l15][0];
#pragma unroll
      for (int r = 0; r < 8; ++r)
        pf.u[r] = prow[((r & 4) ? 8 : 0) + (r & 3) + hi * 4];
    }
    // ---- O += P * V  (V transposed: K-pairs contiguous dwords) --------
#pragma unroll
    for (int dt = 0; dt < 8; ++dt) {
      int d = dt * 16 + l15;
      const unsigned* vrow =
          (const unsigned*)(vt + ((size_t)kvh * 128 + d) * 2048 + j * 32);
      FragAB vf;
#pragma unroll
      for (int r = 0; r < 8; ++r) vf.u[r] = vrow[((r & 4) ? 8 : 0) + (r & 3) + hi * 4];
      o[dt].v = __builtin_amdgcn_wmma_f32_16x16x32_bf16(
          false, pf.v, false, vf.v, (short)0, o[dt].v, false, false);
    }
  }

  // ---- epilogue: normalize, write [s][h*128+d] bf16 -------------------
#pragma unroll
  for (int dt = 0; dt < 8; ++dt)
#pragma unroll
    for (int e = 0; e < 8; ++e) {
      int srow = qrow_base + e + hi * 8;
      int col  = h * 128 + dt * 16 + l15;
      out[(size_t)srow * 4096 + col] = f2bf(o[dt].f[e] / run_sum[e]);
    }
}

// ---------------------------------------------------------------------------
// Host launcher
// ---------------------------------------------------------------------------
extern "C" void kernel_launch(void* const* d_in, const int* in_sizes, int n_in,
                              void* d_out, int out_size, void* d_ws, size_t ws_size,
                              hipStream_t stream) {
  (void)in_sizes; (void)n_in; (void)out_size; (void)ws_size;
  constexpr int S = 2048, H = 4096, NH = 32, NKV = 8, HD = 128;

  const float* hidden = (const float*)d_in[0];
  const float* cosT   = (const float*)d_in[1];
  const float* sinT   = (const float*)d_in[2];
  /* d_in[3] attention_mask: causal, reconstructed analytically */
  const float* Wq     = (const float*)d_in[4];
  const float* Wk     = (const float*)d_in[5];
  const float* Wv     = (const float*)d_in[6];
  const float* Wo     = (const float*)d_in[7];
  const float* sinks  = (const float*)d_in[8];
  float* out = (float*)d_out;

  char* ws = (char*)d_ws;
  size_t off = 0;
  auto alloc = [&](size_t bytes) -> char* {
    char* p = ws + off;
    off += (bytes + 255) & ~(size_t)255;
    return p;
  };

  __bf16* hid_bf = (__bf16*)alloc((size_t)S * H * 2);
  __bf16* wq_bf  = (__bf16*)alloc((size_t)H * (NH * HD) * 2);
  __bf16* wk_bf  = (__bf16*)alloc((size_t)H * (NKV * HD) * 2);
  __bf16* wv_bf  = (__bf16*)alloc((size_t)H * (NKV * HD) * 2);
  __bf16* wo_bf  = (__bf16*)alloc((size_t)(NH * HD) * H * 2);
  float*  qf32   = (float*) alloc((size_t)S * (NH * HD) * 4);
  float*  kf32   = (float*) alloc((size_t)S * (NKV * HD) * 4);
  float*  vf32   = (float*) alloc((size_t)S * (NKV * HD) * 4);
  __bf16* q_bf   = (__bf16*)alloc((size_t)NH * S * HD * 2);
  __bf16* k_bf   = (__bf16*)alloc((size_t)NKV * S * HD * 2);
  __bf16* vT_bf  = (__bf16*)alloc((size_t)NKV * HD * S * 2);
  __bf16* at_bf  = (__bf16*)alloc((size_t)S * (NH * HD) * 2);

  // 1) convert inputs to bf16
  cvt_bf16_kernel<<<8192, 256, 0, stream>>>(hidden, hid_bf, S * H);
  cvt_bf16_kernel<<<8192, 256, 0, stream>>>(Wq, wq_bf, H * NH * HD);
  cvt_bf16_kernel<<<8192, 256, 0, stream>>>(Wk, wk_bf, H * NKV * HD);
  cvt_bf16_kernel<<<8192, 256, 0, stream>>>(Wv, wv_bf, H * NKV * HD);
  cvt_bf16_kernel<<<8192, 256, 0, stream>>>(Wo, wo_bf, NH * HD * H);

  // 2) QKV projections (WMMA)
  gemm_bf16_kernel<4096, 4096><<<dim3(32, 16), 256, 0, stream>>>(hid_bf, wq_bf, qf32);
  gemm_bf16_kernel<1024, 4096><<<dim3(8, 16), 256, 0, stream>>>(hid_bf, wk_bf, kf32);
  gemm_bf16_kernel<1024, 4096><<<dim3(8, 16), 256, 0, stream>>>(hid_bf, wv_bf, vf32);

  // 3) RoPE + packing (scaling HD^-0.5 folded into Q)
  const float scaling = 0.08838834764831845f;  // 128^-0.5
  pack_rope_kernel<<<8192, 256, 0, stream>>>(qf32, cosT, sinT, q_bf, NH, scaling);
  pack_rope_kernel<<<4096, 256, 0, stream>>>(kf32, cosT, sinT, k_bf, NKV, 1.0f);
  pack_vT_kernel<<<4096, 256, 0, stream>>>(vf32, vT_bf);

  // 4) flash attention with sink: 4096 waves = 512 blocks x 8 waves
  flash_attn_kernel<<<512, 256, 0, stream>>>(q_bf, k_bf, vT_bf, sinks, at_bf);

  // 5) output projection (WMMA) -> f32 d_out
  gemm_bf16_kernel<4096, 4096><<<dim3(32, 16), 256, 0, stream>>>(at_bf, wo_bf, out);
}
